// SelfAttention_74517682585981
// MI455X (gfx1250) — matrix operand order repaired and checked by reading, not compiled, not run on hardware
//
#include <hip/hip_runtime.h>
#include <math.h>

#define T_TOK   4100
#define HIDDEN  4096
#define NHEAD   32
#define DHEAD   128
#define SEQ     1024
#define N_PRE   4
#define N_DEC   4
#define NQKV    (3 * HIDDEN)             // 12288
#define ATT_SCALE 0.08838834764831845f   // 128^-0.5

typedef unsigned short u16;
typedef __attribute__((ext_vector_type(16))) __bf16 v16bf;
typedef __attribute__((ext_vector_type(8)))  __bf16 v8bf;
typedef __attribute__((ext_vector_type(8)))  float  v8f;

// ---------------- bf16 helpers (bit-level, RNE) ----------------
__device__ __forceinline__ u16 f2bf(float f) {
    union { float f; unsigned u; } v; v.f = f;
    unsigned u = v.u;
    if ((u & 0x7F800000u) == 0x7F800000u) {           // Inf/NaN: keep class
        unsigned r = u >> 16;
        if (u & 0xFFFFu) r |= 0x40u;                  // quiet any NaN payload
        return (u16)r;
    }
    return (u16)((u + 0x7FFFu + ((u >> 16) & 1u)) >> 16);
}
__device__ __forceinline__ float bf2f(u16 h) {
    union { unsigned u; float f; } v; v.u = ((unsigned)h) << 16;
    return v.f;
}
__device__ __forceinline__ __bf16 f2bfv(float f) {
    union { u16 u; __bf16 b; } v; v.u = f2bf(f); return v.b;
}
__device__ __forceinline__ v8bf bzero8() {
    v8bf z;
#pragma unroll
    for (int i = 0; i < 8; ++i) z[i] = (__bf16)0.0f;
    return z;
}
__device__ __forceinline__ v8f fzero8() {
    v8f z;
#pragma unroll
    for (int i = 0; i < 8; ++i) z[i] = 0.0f;
    return z;
}

// ---------------- WMMA wrapper ----------------
__device__ __forceinline__ v8f wmma_bf16(v16bf a, v16bf b, v8f c) {
    // D = A(16x32 bf16) x B(32x16 bf16) + C(16x16 f32)
    return __builtin_amdgcn_wmma_f32_16x16x32_bf16(false, a, false, b, (short)0, c,
                                                   false, false);
}

// A-matrix fragment (16x32 bf16), ISA interleaved layout:
// lane<16: row=lane, K = kbase+[0..7], kbase+16+[0..7]
// lane>=16: row=lane-16, K = kbase+8+[0..7], kbase+24+[0..7]
__device__ __forceinline__ v16bf frag_a(const u16* base, int stride, int row0,
                                        int kbase, int lane) {
    int m  = lane & 15;
    int hi = (lane >> 4) & 1;
    const u16* p = base + (size_t)(row0 + m) * stride + kbase + hi * 8;
    v8bf lo = *(const v8bf*)p;
    v8bf h2 = *(const v8bf*)(p + 16);
    v16bf r;
#pragma unroll
    for (int i = 0; i < 8; ++i) { r[i] = lo[i]; r[i + 8] = h2[i]; }
    return r;
}

// B-matrix fragment (32x16 bf16) from N-major (NxK) storage:
// lane<16: col n=lane, K = kbase+[0..7], kbase+16+[0..7];  lane>=16: n=lane-16, K = kbase+8+[0..7], kbase+24+[0..7]
__device__ __forceinline__ v16bf frag_b(const u16* base, int stride, int row0,
                                        int kbase, int lane) {
    int n  = lane & 15;
    int hi = (lane >> 4) & 1;
    const u16* p = base + (size_t)(row0 + n) * stride + kbase + hi * 8;
    v8bf lo = *(const v8bf*)p;
    v8bf h2 = *(const v8bf*)(p + 16);
    v16bf r;
#pragma unroll
    for (int i = 0; i < 8; ++i) { r[i] = lo[i]; r[i + 8] = h2[i]; }
    return r;
}

// ---------------- conversion / transpose kernels ----------------
__global__ void f32_to_bf16_kernel(const float* __restrict__ in, u16* __restrict__ out,
                                   long n) {
    long i = (long)blockIdx.x * blockDim.x + threadIdx.x;
    if (i < n) out[i] = f2bf(in[i]);
}

// w is K x N (row major fp32); wT is N x K bf16
__global__ __launch_bounds__(256) void transpose_to_bf16(const float* __restrict__ w,
                                                         u16* __restrict__ wT,
                                                         int K, int N) {
    __shared__ float tile[32][33];
    int n0 = blockIdx.x * 32, k0 = blockIdx.y * 32;
    for (int i = threadIdx.y; i < 32; i += 8) {
        int k = k0 + i, n = n0 + threadIdx.x;
        tile[i][threadIdx.x] = w[(size_t)k * N + n];
    }
    __syncthreads();
    for (int i = threadIdx.y; i < 32; i += 8) {
        int n = n0 + i, k = k0 + threadIdx.x;
        wT[(size_t)n * K + k] = f2bf(tile[threadIdx.x][i]);
    }
}

// ---------------- WMMA GEMM: C[MxN] = A[MxK] * BT[NxK]^T ----------------
// Block tile 128(M) x 256(N), 8 waves (2x4), wave tile 64x64 (16 wmma / K-step).
// Double-buffered LDS + register prefetch: one barrier per 32-wide K-step,
// global loads for step k+1 overlap the 16 WMMAs of step k.
template <bool OUT_BF16>
__global__ __launch_bounds__(256) void gemm_wmma(const u16* __restrict__ A,
                                                 const u16* __restrict__ BT,
                                                 void* __restrict__ Cout,
                                                 int M, int N, int K) {
    __shared__ u16 lA[2][128][40];   // 20 KB
    __shared__ u16 lB[2][256][40];   // 40 KB
    const int tid  = threadIdx.x;
    const int lane = tid & 31;
    const int wave = tid >> 5;
    const int wm = wave >> 2;        // 0..1 -> 64 rows
    const int wn = wave & 3;         // 0..3 -> 64 cols
    const int bm = blockIdx.y * 128;
    const int bn = blockIdx.x * 256;

    const int ldrowA = tid >> 2;           // 0..63 base row (A loads: +64 per i)
    const int ldc8   = (tid & 3) << 3;     // 0,8,16,24

    v8f acc[4][4];
#pragma unroll
    for (int mt = 0; mt < 4; ++mt)
#pragma unroll
        for (int nt = 0; nt < 4; ++nt) acc[mt][nt] = fzero8();

    const int nk = K >> 5;

    // ---- prologue: stage 0 into buffer 0 ----
    {
#pragma unroll
        for (int i = 0; i < 2; ++i) {
            int row = ldrowA + i * 64;
            int gm = bm + row;
            v8bf va = bzero8();
            if (gm < M) va = *(const v8bf*)(A + (size_t)gm * K + ldc8);
            *(v8bf*)&lA[0][row][ldc8] = va;
        }
#pragma unroll
        for (int i = 0; i < 4; ++i) {
            int row = ldrowA + i * 64;
            int gn = bn + row;                       // N % 256 == 0: in range
            *(v8bf*)&lB[0][row][ldc8] = *(const v8bf*)(BT + (size_t)gn * K + ldc8);
        }
    }
    __syncthreads();

    for (int kc = 0; kc < nk; ++kc) {
        const int cur = kc & 1;
        const bool more = (kc + 1) < nk;
        v8bf pa[2], pb[4];
        if (more) {                                   // prefetch next K-step
            const int k0 = (kc + 1) << 5;
#pragma unroll
            for (int i = 0; i < 2; ++i) {
                int row = ldrowA + i * 64;
                int gm = bm + row;
                pa[i] = bzero8();
                if (gm < M) pa[i] = *(const v8bf*)(A + (size_t)gm * K + k0 + ldc8);
            }
#pragma unroll
            for (int i = 0; i < 4; ++i) {
                int row = ldrowA + i * 64;
                int gn = bn + row;
                pb[i] = *(const v8bf*)(BT + (size_t)gn * K + k0 + ldc8);
            }
        }
        // ---- compute current buffer: 16 WMMAs ----
        {
            v16bf af[4], bfr[4];
#pragma unroll
            for (int mt = 0; mt < 4; ++mt)
                af[mt] = frag_a(&lA[cur][0][0], 40, wm * 64 + mt * 16, 0, lane);
#pragma unroll
            for (int nt = 0; nt < 4; ++nt)
                bfr[nt] = frag_b(&lB[cur][0][0], 40, wn * 64 + nt * 16, 0, lane);
#pragma unroll
            for (int mt = 0; mt < 4; ++mt)
#pragma unroll
                for (int nt = 0; nt < 4; ++nt)
                    acc[mt][nt] = wmma_bf16(af[mt], bfr[nt], acc[mt][nt]);
        }
        if (more) {                                   // spill prefetch into other buf
            const int nxt = cur ^ 1;
#pragma unroll
            for (int i = 0; i < 2; ++i)
                *(v8bf*)&lA[nxt][ldrowA + i * 64][ldc8] = pa[i];
#pragma unroll
            for (int i = 0; i < 4; ++i)
                *(v8bf*)&lB[nxt][ldrowA + i * 64][ldc8] = pb[i];
        }
        __syncthreads();
    }

    const int hi = (lane >> 4) & 1;
#pragma unroll
    for (int mt = 0; mt < 4; ++mt)
#pragma unroll
        for (int nt = 0; nt < 4; ++nt)
#pragma unroll
            for (int r = 0; r < 8; ++r) {
                int gm = bm + wm * 64 + mt * 16 + r + hi * 8;
                int gn = bn + wn * 64 + nt * 16 + (lane & 15);
                if (gm < M) {
                    if (OUT_BF16)
                        ((u16*)Cout)[(size_t)gm * N + gn] = f2bf(acc[mt][nt][r]);
                    else
                        ((float*)Cout)[(size_t)gm * N + gn] = acc[mt][nt][r];
                }
            }
}

// ---------------- RoPE: qkv bf16 -> q_rope, k_rope bf16 ----------------
__global__ void rope_kernel(const u16* __restrict__ qkvb, const int* __restrict__ pos,
                            u16* __restrict__ qr, u16* __restrict__ kr) {
    long idx = (long)blockIdx.x * blockDim.x + threadIdx.x;
    if (idx >= (long)T_TOK * NHEAD * 64) return;
    int d = (int)(idx & 63);
    int h = (int)((idx >> 6) & 31);
    int t = (int)(idx >> 11);
    float p = (float)pos[t];
    float invf = __powf(10000.0f, -(float)d * (1.0f / 64.0f));
    float ang = p * invf;
    float c = __cosf(ang), s = __sinf(ang);
    size_t base = (size_t)t * NQKV + h * DHEAD + d;
    size_t ob   = (size_t)t * HIDDEN + h * DHEAD + d;
    float q1 = bf2f(qkvb[base]), q2 = bf2f(qkvb[base + 64]);
    qr[ob]      = f2bf(q1 * c - q2 * s);
    qr[ob + 64] = f2bf(q2 * c + q1 * s);
    float k1 = bf2f(qkvb[base + HIDDEN]), k2 = bf2f(qkvb[base + HIDDEN + 64]);
    kr[ob]      = f2bf(k1 * c - k2 * s);
    kr[ob + 64] = f2bf(k2 * c + k1 * s);
}

// ---------------- Flash-attention prefill (WMMA) ----------------
// Block: 4 waves, each wave owns one 16-query tile of (seq, head).
// Per 32-key chunk: S^T = K*Q^T (8 wmma), softmax along VGPRs, O += P*V (8 wmma).
__global__ __launch_bounds__(128) void attn_prefill(const u16* __restrict__ qr,
                                                    const u16* __restrict__ kr,
                                                    const u16* __restrict__ qkvb,
                                                    u16* __restrict__ ao) {
    __shared__ u16 lK[32][136];   // [key][d], pad 8
    __shared__ u16 lVt[128][40];  // [d][key], pad 8
    const int lane = threadIdx.x & 31;
    const int wave = threadIdx.x >> 5;
    const int hi   = (lane >> 4) & 1;
    const int qcol = lane & 15;
    const int head = blockIdx.y;
    const int seq_base = blockIdx.z * SEQ;
    const int qtile = blockIdx.x * 4 + wave;

    // Q fragments (B layout), constant across key chunks: 4 K-steps over D=128
    v16bf bq[4];
    {
        const u16* qbase = qr + (size_t)(seq_base + qtile * 16 + qcol) * HIDDEN +
                           head * DHEAD;
#pragma unroll
        for (int kk = 0; kk < 4; ++kk) {
            const u16* p = qbase + kk * 32 + hi * 8;
            v8bf lo = *(const v8bf*)p;
            v8bf h2 = *(const v8bf*)(p + 16);
#pragma unroll
            for (int i = 0; i < 8; ++i) { bq[kk][i] = lo[i]; bq[kk][i + 8] = h2[i]; }
        }
    }

    v8f o[8];
#pragma unroll
    for (int ds = 0; ds < 8; ++ds) o[ds] = fzero8();
    float m = -INFINITY, l = 0.0f;

    const int nch_w = (qtile * 16 + 15) / 32 + 1;
    const int nch_b = (blockIdx.x * 64 + 63) / 32 + 1;

    for (int c = 0; c < nch_b; ++c) {
        // cooperative loads: K chunk 32x128, V chunk transposed to [d][key]
        for (int i = threadIdx.x; i < 512; i += 128) {
            int row = i >> 4, c8 = (i & 15) << 3;
            const u16* src = kr + (size_t)(seq_base + c * 32 + row) * HIDDEN +
                             head * DHEAD + c8;
            *(v8bf*)&lK[row][c8] = *(const v8bf*)src;
        }
        for (int i = threadIdx.x; i < 32 * 128; i += 128) {
            int key = i >> 7, d = i & 127;
            lVt[d][key] = qkvb[(size_t)(seq_base + c * 32 + key) * NQKV +
                               2 * HIDDEN + head * DHEAD + d];
        }
        __syncthreads();

        if (c < nch_w) {
            v8f st0 = fzero8(), st1 = fzero8();
#pragma unroll
            for (int kk = 0; kk < 4; ++kk) {
                v16bf a0 = frag_a(&lK[0][0], 136, 0, kk * 32, lane);
                v16bf a1 = frag_a(&lK[0][0], 136, 16, kk * 32, lane);
                st0 = wmma_bf16(a0, bq[kk], st0);
                st1 = wmma_bf16(a1, bq[kk], st1);
            }
            const int qg = qtile * 16 + qcol;
            float s0[8], s1[8], mloc = -INFINITY;
#pragma unroll
            for (int r = 0; r < 8; ++r) {
                int key0 = c * 32 + r + hi * 8;
                s0[r] = (key0 <= qg) ? st0[r] * ATT_SCALE : -INFINITY;
                s1[r] = (key0 + 16 <= qg) ? st1[r] * ATT_SCALE : -INFINITY;
                mloc = fmaxf(mloc, fmaxf(s0[r], s1[r]));
            }
            mloc = fmaxf(mloc, __shfl_xor(mloc, 16, 32));
            float mn = fmaxf(m, mloc);
            float p0[8], p1[8], ls = 0.0f;
#pragma unroll
            for (int r = 0; r < 8; ++r) {
                p0[r] = __expf(s0[r] - mn);
                p1[r] = __expf(s1[r] - mn);
                ls += p0[r] + p1[r];
            }
            ls += __shfl_xor(ls, 16, 32);
            float alpha = __expf(m - mn);
            l = l * alpha + ls;
            m = mn;
            float aB[8];
#pragma unroll
            for (int r = 0; r < 8; ++r) aB[r] = __shfl(alpha, r + hi * 8, 32);
#pragma unroll
            for (int ds = 0; ds < 8; ++ds)
#pragma unroll
                for (int r = 0; r < 8; ++r) o[ds][r] *= aB[r];
            // S^T C-layout packs directly into P's A-fragment layout
            v16bf pa;
#pragma unroll
            for (int i = 0; i < 8; ++i) {
                pa[i] = f2bfv(p0[i]);
                pa[i + 8] = f2bfv(p1[i]);
            }
#pragma unroll
            for (int ds = 0; ds < 8; ++ds) {
                v16bf bv = frag_b(&lVt[0][0], 40, ds * 16, 0, lane);
                o[ds] = wmma_bf16(pa, bv, o[ds]);
            }
        }
        __syncthreads();
    }

    float linv = 1.0f / l;
    float lB[8];
#pragma unroll
    for (int r = 0; r < 8; ++r) lB[r] = __shfl(linv, r + hi * 8, 32);
#pragma unroll
    for (int ds = 0; ds < 8; ++ds)
#pragma unroll
        for (int r = 0; r < 8; ++r) {
            int t = seq_base + qtile * 16 + r + hi * 8;
            int col = head * DHEAD + ds * 16 + qcol;
            ao[(size_t)t * HIDDEN + col] = f2bf(o[ds][r] * lB[r]);
        }
}

// ---------------- Decode attention (4 tokens; negligible FLOPs) ----------------
__global__ __launch_bounds__(128) void attn_decode(const u16* __restrict__ qr,
                                                   const u16* __restrict__ kr,
                                                   const u16* __restrict__ qkvb,
                                                   u16* __restrict__ ao) {
    __shared__ float sc[SEQ];
    __shared__ float red[128];
    __shared__ float qv[DHEAD];
    const int tid = threadIdx.x;
    const int h = blockIdx.x;
    const int t = N_PRE * SEQ + blockIdx.y;
    qv[tid] = bf2f(qr[(size_t)t * HIDDEN + h * DHEAD + tid]);
    __syncthreads();
    for (int k = tid; k < SEQ; k += 128) {
        const u16* kp = kr + (size_t)k * HIDDEN + h * DHEAD;
        float s = 0.0f;
        for (int d = 0; d < DHEAD; ++d) s += bf2f(kp[d]) * qv[d];
        sc[k] = s * ATT_SCALE;
    }
    __syncthreads();
    float mloc = -INFINITY;
    for (int k = tid; k < SEQ; k += 128) mloc = fmaxf(mloc, sc[k]);
    red[tid] = mloc; __syncthreads();
    for (int s = 64; s > 0; s >>= 1) {
        if (tid < s) red[tid] = fmaxf(red[tid], red[tid + s]);
        __syncthreads();
    }
    float mm = red[0]; __syncthreads();
    float ls = 0.0f;
    for (int k = tid; k < SEQ; k += 128) {
        float p = __expf(sc[k] - mm);
        sc[k] = p;
        ls += p;
    }
    red[tid] = ls; __syncthreads();
    for (int s = 64; s > 0; s >>= 1) {
        if (tid < s) red[tid] += red[tid + s];
        __syncthreads();
    }
    float L = red[0];
    float acc = 0.0f;
    for (int k = 0; k < SEQ; ++k)
        acc += sc[k] * bf2f(qkvb[(size_t)k * NQKV + 2 * HIDDEN + h * DHEAD + tid]);
    ao[(size_t)t * HIDDEN + h * DHEAD + tid] = f2bf(acc / L);
}

// ---------------- launch ----------------
extern "C" void kernel_launch(void* const* d_in, const int* in_sizes, int n_in,
                              void* d_out, int out_size, void* d_ws, size_t ws_size,
                              hipStream_t stream) {
    (void)in_sizes; (void)n_in; (void)out_size; (void)ws_size;
    const float* x     = (const float*)d_in[0];
    const float* w_qkv = (const float*)d_in[3];
    const float* w_out = (const float*)d_in[4];
    const int*   pos   = (const int*)d_in[5];

    char* ws = (char*)d_ws;
    size_t off = 0;
    auto alloc = [&](size_t bytes) {
        void* p = ws + off;
        off += (bytes + 255) & ~(size_t)255;
        return p;
    };
    u16* xb    = (u16*)alloc((size_t)T_TOK * HIDDEN * 2);
    u16* wqkvT = (u16*)alloc((size_t)NQKV * HIDDEN * 2);
    u16* woutT = (u16*)alloc((size_t)HIDDEN * HIDDEN * 2);
    u16* qkvb  = (u16*)alloc((size_t)T_TOK * NQKV * 2);
    u16* qrope = (u16*)alloc((size_t)T_TOK * HIDDEN * 2);
    u16* krope = (u16*)alloc((size_t)T_TOK * HIDDEN * 2);
    u16* ao    = (u16*)alloc((size_t)T_TOK * HIDDEN * 2);

    {
        long n = (long)T_TOK * HIDDEN;
        f32_to_bf16_kernel<<<(unsigned)((n + 255) / 256), 256, 0, stream>>>(x, xb, n);
    }
    transpose_to_bf16<<<dim3(NQKV / 32, HIDDEN / 32), dim3(32, 8), 0, stream>>>(
        w_qkv, wqkvT, HIDDEN, NQKV);
    transpose_to_bf16<<<dim3(HIDDEN / 32, HIDDEN / 32), dim3(32, 8), 0, stream>>>(
        w_out, woutT, HIDDEN, HIDDEN);
    gemm_wmma<true><<<dim3(NQKV / 256, (T_TOK + 127) / 128), 256, 0, stream>>>(
        xb, wqkvT, qkvb, T_TOK, NQKV, HIDDEN);
    {
        long n = (long)T_TOK * NHEAD * 64;
        rope_kernel<<<(unsigned)((n + 255) / 256), 256, 0, stream>>>(qkvb, pos, qrope,
                                                                     krope);
    }
    attn_prefill<<<dim3(SEQ / 64, NHEAD, N_PRE), 128, 0, stream>>>(qrope, krope, qkvb,
                                                                   ao);
    attn_decode<<<dim3(NHEAD, N_DEC), 128, 0, stream>>>(qrope, krope, qkvb, ao);
    gemm_wmma<false><<<dim3(HIDDEN / 256, (T_TOK + 127) / 128), 256, 0, stream>>>(
        ao, woutT, d_out, T_TOK, HIDDEN, HIDDEN);
}
